// MIPNetwork_87875030877000
// MI455X (gfx1250) — matrix operand order repaired and checked by reading, not compiled, not run on hardware
//
#include <hip/hip_runtime.h>
#include <math.h>

#define FM 64
#define OUT_BITS 16
#define PASS_STEPS 3

typedef __attribute__((ext_vector_type(16))) __bf16         v16bf;
typedef __attribute__((ext_vector_type(8)))  float          v8f;
typedef __attribute__((ext_vector_type(8)))  unsigned short v8us;

union frag16 { v16bf bf; v8us us2[2]; };

__device__ __forceinline__ unsigned short f32_to_bf16(float f) {
  union { float f; unsigned int u; } c; c.f = f;
  unsigned int u = c.u;
  return (unsigned short)((u + 0x7FFFu + ((u >> 16) & 1u)) >> 16);  // RNE
}

// ---- WMMA fragment helpers (layouts per CDNA5 ISA §7.12.2) ----

// A: 16x32 bf16 from row-major bf16 LDS buffer, row stride LD elements.
template<int LD>
__device__ __forceinline__ v16bf load_a_lds(const unsigned short* buf, int lane, int k0) {
  const unsigned short* p = buf + (lane & 15) * LD + k0 + ((lane >> 4) << 3);
  frag16 f;
  f.us2[0] = *(const v8us*)(p);        // K = base+0..7
  f.us2[1] = *(const v8us*)(p + 16);   // K = base+16..23
  return f.bf;
}

// B: 32x16 bf16 from weights pre-transposed to [N][K] (ldk elems).
__device__ __forceinline__ v16bf load_b_wt(const unsigned short* WT, int ldk,
                                           int lane, int k0, int n0) {
  const unsigned short* p = WT + (size_t)(n0 + (lane & 15)) * ldk + k0 + ((lane >> 4) << 3);
  frag16 f;
  f.us2[0] = *(const v8us*)(p);
  f.us2[1] = *(const v8us*)(p + 16);
  return f.bf;
}

__device__ __forceinline__ v8f init_acc(const float* b, int n) {
  float bv = b[n];
  v8f a;
  #pragma unroll
  for (int i = 0; i < 8; ++i) a[i] = bv;
  return a;
}

__device__ __forceinline__ v8f wmma_bf16(v16bf a, v16bf b, v8f c) {
  return __builtin_amdgcn_wmma_f32_16x16x32_bf16(false, a, false, b, (short)0, c,
                                                 false, false);
}

// Async copy of one 16-byte chunk (global bf16 -> LDS), tracked by ASYNCcnt.
__device__ __forceinline__ void async_b128(unsigned lds_addr, const void* gptr) {
  unsigned long long ga = (unsigned long long)(size_t)gptr;
  asm volatile("global_load_async_to_lds_b128 %0, %1, off"
               :: "v"(lds_addr), "v"(ga) : "memory");
}
__device__ __forceinline__ void wait_async0() {
  asm volatile("s_wait_asynccnt 0x0" ::: "memory");
}

// ---------------- utility kernels ----------------

__global__ void k_fill_f32(float* __restrict__ p, long long n, float v) {
  long long i = (long long)blockIdx.x * blockDim.x + threadIdx.x;
  if (i < n) p[i] = v;
}
__global__ void k_fill_us(unsigned short* __restrict__ p, long long n, unsigned short v) {
  long long i = (long long)blockIdx.x * blockDim.x + threadIdx.x;
  if (i < n) p[i] = v;
}
__global__ void k_fill_i32(int* __restrict__ p, long long n, int v) {
  long long i = (long long)blockIdx.x * blockDim.x + threadIdx.x;
  if (i < n) p[i] = v;
}
__global__ void k_copy_i32(const int* __restrict__ s, int* __restrict__ d, int n) {
  int i = blockIdx.x * blockDim.x + threadIdx.x;
  if (i < n) d[i] = s[i];
}

__global__ void k_transpose_bf16(const float* __restrict__ W, unsigned short* __restrict__ WT,
                                 int K, int N) {
  int i = blockIdx.x * blockDim.x + threadIdx.x;
  if (i >= K * N) return;
  int k = i / N, n = i - k * N;
  WT[(size_t)n * K + k] = f32_to_bf16(W[(size_t)k * N + n]);
}

// ---------------- CSR build (counting sort by destination) ----------------

__global__ void k_count(const int* __restrict__ dst, int* __restrict__ cnt, int nnz) {
  int e = blockIdx.x * blockDim.x + threadIdx.x;
  if (e < nnz) atomicAdd(&cnt[dst[e]], 1);
}

// single-block exclusive scan: ptr[0..N-1] = exclusive prefix, ptr[N] = total
__global__ void __launch_bounds__(256)
k_scan_excl(const int* __restrict__ cnt, int* __restrict__ ptr, int N) {
  __shared__ int s[256];
  __shared__ int carry;
  if (threadIdx.x == 0) carry = 0;
  __syncthreads();
  for (int base = 0; base < N; base += 256) {
    int i = base + threadIdx.x;
    int v = (i < N) ? cnt[i] : 0;
    s[threadIdx.x] = v;
    __syncthreads();
    #pragma unroll
    for (int off = 1; off < 256; off <<= 1) {
      int t = (threadIdx.x >= off) ? s[threadIdx.x - off] : 0;
      __syncthreads();
      s[threadIdx.x] += t;
      __syncthreads();
    }
    int c = carry;
    if (i < N) ptr[i] = c + s[threadIdx.x] - v;
    __syncthreads();
    if (threadIdx.x == 0) carry = c + s[255];
    __syncthreads();
  }
  if (threadIdx.x == 0) ptr[N] = carry;
}

__global__ void k_scatter(const int* __restrict__ dst, int* __restrict__ fill,
                          int* __restrict__ perm, int nnz) {
  int e = blockIdx.x * blockDim.x + threadIdx.x;
  if (e < nnz) {
    int p = atomicAdd(&fill[dst[e]], 1);
    perm[p] = e;
  }
}

// ---------------- gather-side SpMM (no atomics in hot loop) ----------------
// Yb[n,:] = bf16( sum_{j in [ptr[n],ptr[n+1])} |vals[perm[j]]| * X[other[perm[j]],:] )
__global__ void __launch_bounds__(256)
k_spmm_csr(const int* __restrict__ rowptr, const int* __restrict__ perm,
           const int* __restrict__ other_idx, const float* __restrict__ vals,
           const float* __restrict__ X, unsigned short* __restrict__ Yb, int N) {
  int wave = threadIdx.x >> 5, lane = threadIdx.x & 31;
  int n = blockIdx.x * 8 + wave;
  if (n >= N) return;
  int beg = rowptr[n], end = rowptr[n + 1];
  float a0 = 0.f, a1 = 0.f;
  for (int j = beg; j < end; ++j) {
    int e = perm[j];
    int o = other_idx[e];
    float v = fabsf(vals[e]);
    const float* xr = X + (size_t)o * FM;
    a0 = fmaf(v, xr[lane], a0);
    a1 = fmaf(v, xr[lane + 32], a1);
  }
  Yb[(size_t)n * FM + lane]      = f32_to_bf16(a0);
  Yb[(size_t)n * FM + lane + 32] = f32_to_bf16(a1);
}

// ---------------- fused WMMA MLP kernels ----------------
// block = 128 threads = 4 waves; wave w owns 16 output rows.

// constraints = (relu(cond*w1+b1)) @ W2 + b2  (layer1 elementwise: K_in = 1)
__global__ void __launch_bounds__(128)
k_prepare_cond(const float* __restrict__ cond,
               const float* __restrict__ w1, const float* __restrict__ b1,   // [128]
               const unsigned short* __restrict__ W2T, const float* __restrict__ b2,
               float* __restrict__ OUTf, unsigned short* __restrict__ OUTb, int M) {
  __shared__ unsigned short sH[4][16 * 128];
  int wave = threadIdx.x >> 5, lane = threadIdx.x & 31;
  int row0 = (blockIdx.x * 4 + wave) * 16;
  __builtin_prefetch(W2T, 0, 0);

  for (int i = 0; i < 64; ++i) {                 // 16 rows x 128 cols
    int idx = i * 32 + lane;
    int r = idx >> 7, j = idx & 127;
    int gr = row0 + r; if (gr > M - 1) gr = M - 1;
    float v = cond[gr] * w1[j] + b1[j];
    sH[wave][r * 128 + j] = f32_to_bf16(v > 0.f ? v : 0.f);
  }
  __syncthreads();

  v8f acc[4];
  #pragma unroll
  for (int nt = 0; nt < 4; ++nt) acc[nt] = init_acc(b2, nt * 16 + (lane & 15));
  #pragma unroll
  for (int k0 = 0; k0 < 128; k0 += 32) {
    v16bf a = load_a_lds<128>(sH[wave], lane, k0);
    #pragma unroll
    for (int nt = 0; nt < 4; ++nt)
      acc[nt] = wmma_bf16(a, load_b_wt(W2T, 128, lane, k0, nt * 16), acc[nt]);
  }
  #pragma unroll
  for (int nt = 0; nt < 4; ++nt) {
    int n = nt * 16 + (lane & 15);
    #pragma unroll
    for (int e = 0; e < 8; ++e) {
      int gr = row0 + ((lane >> 4) << 3) + e;
      if (gr < M) {
        OUTf[(size_t)gr * FM + n] = acc[nt][e];
        OUTb[(size_t)gr * FM + n] = f32_to_bf16(acc[nt][e]);
      }
    }
  }
}

// OUT = relu(concat(X1b,X2b)[M,128] @ W1 + b1) @ W2 + b2 ; writes f32 + bf16 mirror.
// Safe for OUTb == X1b (each block touches only its own rows).
__global__ void __launch_bounds__(128)
k_mlp2_concat(const unsigned short* __restrict__ X1b, const unsigned short* __restrict__ X2b,
              const unsigned short* __restrict__ W1T, const float* __restrict__ b1,
              const unsigned short* __restrict__ W2T, const float* __restrict__ b2,
              float* __restrict__ OUTf, unsigned short* __restrict__ OUTb, int M) {
  __shared__ unsigned short sA[4][16 * 128];
  __shared__ unsigned short sH[4][16 * 128];
  int wave = threadIdx.x >> 5, lane = threadIdx.x & 31;
  int row0 = (blockIdx.x * 4 + wave) * 16;
  __builtin_prefetch(W1T, 0, 0);

  // async-stage concat(X1b,X2b) -> sA : 16 rows x 16 chunks of 16B
  #pragma unroll
  for (int i = 0; i < 8; ++i) {
    int t = i * 32 + lane;
    int r = t >> 4, c16 = t & 15;
    int gr = row0 + r; if (gr > M - 1) gr = M - 1;
    const unsigned short* gp = (c16 < 8)
        ? (X1b + (size_t)gr * FM + c16 * 8)
        : (X2b + (size_t)gr * FM + (c16 - 8) * 8);
    async_b128((unsigned)(size_t)&sA[wave][r * 128 + c16 * 8], gp);
  }
  wait_async0();
  __syncthreads();

  // layer 1: [16,128] @ [128,128] + b1, relu -> sH
  v8f acc[8];
  #pragma unroll
  for (int nt = 0; nt < 8; ++nt) acc[nt] = init_acc(b1, nt * 16 + (lane & 15));
  #pragma unroll
  for (int k0 = 0; k0 < 128; k0 += 32) {
    v16bf a = load_a_lds<128>(sA[wave], lane, k0);
    #pragma unroll
    for (int nt = 0; nt < 8; ++nt)
      acc[nt] = wmma_bf16(a, load_b_wt(W1T, 128, lane, k0, nt * 16), acc[nt]);
  }
  #pragma unroll
  for (int nt = 0; nt < 8; ++nt) {
    int n = nt * 16 + (lane & 15);
    #pragma unroll
    for (int e = 0; e < 8; ++e) {
      int m = ((lane >> 4) << 3) + e;
      float v = acc[nt][e];
      sH[wave][m * 128 + n] = f32_to_bf16(v > 0.f ? v : 0.f);
    }
  }
  __syncthreads();

  // layer 2: [16,128] @ [128,64] + b2 -> OUT (f32 + bf16 mirror)
  v8f acc2[4];
  #pragma unroll
  for (int nt = 0; nt < 4; ++nt) acc2[nt] = init_acc(b2, nt * 16 + (lane & 15));
  #pragma unroll
  for (int k0 = 0; k0 < 128; k0 += 32) {
    v16bf a = load_a_lds<128>(sH[wave], lane, k0);
    #pragma unroll
    for (int nt = 0; nt < 4; ++nt)
      acc2[nt] = wmma_bf16(a, load_b_wt(W2T, 128, lane, k0, nt * 16), acc2[nt]);
  }
  #pragma unroll
  for (int nt = 0; nt < 4; ++nt) {
    int n = nt * 16 + (lane & 15);
    #pragma unroll
    for (int e = 0; e < 8; ++e) {
      int gr = row0 + ((lane >> 4) << 3) + e;
      if (gr < M) {
        OUTf[(size_t)gr * FM + n] = acc2[nt][e];
        OUTb[(size_t)gr * FM + n] = f32_to_bf16(acc2[nt][e]);
      }
    }
  }
}

// OUT[M,16] = sigmoid(relu(Xb[M,64] @ W1[64,128] + b1) @ W2[128,16] + b2)
__global__ void __launch_bounds__(128)
k_output(const unsigned short* __restrict__ Xb,
         const unsigned short* __restrict__ W1T, const float* __restrict__ b1,
         const unsigned short* __restrict__ W2T, const float* __restrict__ b2,
         float* __restrict__ OUT, int M) {
  __shared__ unsigned short sA[4][16 * 64];
  __shared__ unsigned short sH[4][16 * 128];
  int wave = threadIdx.x >> 5, lane = threadIdx.x & 31;
  int row0 = (blockIdx.x * 4 + wave) * 16;
  __builtin_prefetch(W1T, 0, 0);

  // async-stage Xb rows -> sA : 16 rows x 8 chunks of 16B
  #pragma unroll
  for (int i = 0; i < 4; ++i) {
    int t = i * 32 + lane;
    int r = t >> 3, c8 = t & 7;
    int gr = row0 + r; if (gr > M - 1) gr = M - 1;
    async_b128((unsigned)(size_t)&sA[wave][r * 64 + c8 * 8],
               Xb + (size_t)gr * FM + c8 * 8);
  }
  wait_async0();
  __syncthreads();

  v8f acc[8];
  #pragma unroll
  for (int nt = 0; nt < 8; ++nt) acc[nt] = init_acc(b1, nt * 16 + (lane & 15));
  #pragma unroll
  for (int k0 = 0; k0 < 64; k0 += 32) {
    v16bf a = load_a_lds<64>(sA[wave], lane, k0);
    #pragma unroll
    for (int nt = 0; nt < 8; ++nt)
      acc[nt] = wmma_bf16(a, load_b_wt(W1T, 64, lane, k0, nt * 16), acc[nt]);
  }
  #pragma unroll
  for (int nt = 0; nt < 8; ++nt) {
    int n = nt * 16 + (lane & 15);
    #pragma unroll
    for (int e = 0; e < 8; ++e) {
      int m = ((lane >> 4) << 3) + e;
      float v = acc[nt][e];
      sH[wave][m * 128 + n] = f32_to_bf16(v > 0.f ? v : 0.f);
    }
  }
  __syncthreads();

  v8f acc2 = init_acc(b2, lane & 15);
  #pragma unroll
  for (int k0 = 0; k0 < 128; k0 += 32) {
    v16bf a = load_a_lds<128>(sH[wave], lane, k0);
    acc2 = wmma_bf16(a, load_b_wt(W2T, 128, lane, k0, 0), acc2);
  }
  #pragma unroll
  for (int e = 0; e < 8; ++e) {
    int gr = row0 + ((lane >> 4) << 3) + e;
    if (gr < M) OUT[(size_t)gr * OUT_BITS + (lane & 15)] = 1.f / (1.f + __expf(-acc2[e]));
  }
}

// ---------------- host orchestration ----------------

extern "C" void kernel_launch(void* const* d_in, const int* in_sizes, int n_in,
                              void* d_out, int out_size, void* d_ws, size_t ws_size,
                              hipStream_t stream) {
  const int*   row_idx  = (const int*)d_in[0];
  const int*   col_idx  = (const int*)d_in[1];
  const float* adj_vals = (const float*)d_in[2];
  const float* cond     = (const float*)d_in[3];
  const float* pc_w1 = (const float*)d_in[6];  const float* pc_b1 = (const float*)d_in[7];
  const float* pc_w2 = (const float*)d_in[8];  const float* pc_b2 = (const float*)d_in[9];
  const float* cu_w1 = (const float*)d_in[10]; const float* cu_b1 = (const float*)d_in[11];
  const float* cu_w2 = (const float*)d_in[12]; const float* cu_b2 = (const float*)d_in[13];
  const float* vu_w1 = (const float*)d_in[14]; const float* vu_b1 = (const float*)d_in[15];
  const float* vu_w2 = (const float*)d_in[16]; const float* vu_b2 = (const float*)d_in[17];
  const float* out_w1 = (const float*)d_in[18]; const float* out_b1 = (const float*)d_in[19];
  const float* out_w2 = (const float*)d_in[20]; const float* out_b2 = (const float*)d_in[21];

  const int NNZ = in_sizes[0];
  const int C   = in_sizes[3];
  const int V   = out_size / OUT_BITS;

  char* ws = (char*)d_ws;
  float* vars_f = (float*)ws;           ws += (size_t)V * FM * 4;   // gather source
  float* cons_f = (float*)ws;           ws += (size_t)C * FM * 4;   // gather source
  unsigned short* vars_b = (unsigned short*)ws; ws += (size_t)V * FM * 2;
  unsigned short* cons_b = (unsigned short*)ws; ws += (size_t)C * FM * 2;
  unsigned short* msgc_b = (unsigned short*)ws; ws += (size_t)C * FM * 2;  // v2c
  unsigned short* msgv_b = (unsigned short*)ws; ws += (size_t)V * FM * 2;  // c2v
  unsigned short* pc_w2T  = (unsigned short*)ws; ws += 64  * 128 * 2;
  unsigned short* cu_w1T  = (unsigned short*)ws; ws += 128 * 128 * 2;
  unsigned short* cu_w2T  = (unsigned short*)ws; ws += 64  * 128 * 2;
  unsigned short* vu_w1T  = (unsigned short*)ws; ws += 128 * 128 * 2;
  unsigned short* vu_w2T  = (unsigned short*)ws; ws += 64  * 128 * 2;
  unsigned short* out_w1T = (unsigned short*)ws; ws += 128 * 64  * 2;
  unsigned short* out_w2T = (unsigned short*)ws; ws += 16  * 128 * 2;
  int* cnt_c = (int*)ws;  ws += (size_t)C * 4;        // also reused as fill cursor
  int* cnt_v = (int*)ws;  ws += (size_t)V * 4;
  int* ptr_c = (int*)ws;  ws += (size_t)(C + 1) * 4;
  int* ptr_v = (int*)ws;  ws += (size_t)(V + 1) * 4;
  int* perm_c = (int*)ws; ws += (size_t)NNZ * 4;
  int* perm_v = (int*)ws; ws += (size_t)NNZ * 4;

  const unsigned eBlk = (unsigned)((NNZ + 255) / 256);

  // ---- CSR build (once per launch) ----
  k_fill_i32<<<(unsigned)(((long long)C + 255) / 256), 256, 0, stream>>>(cnt_c, C, 0);
  k_fill_i32<<<(unsigned)(((long long)V + 255) / 256), 256, 0, stream>>>(cnt_v, V, 0);
  k_count<<<eBlk, 256, 0, stream>>>(col_idx, cnt_c, NNZ);
  k_count<<<eBlk, 256, 0, stream>>>(row_idx, cnt_v, NNZ);
  k_scan_excl<<<1, 256, 0, stream>>>(cnt_c, ptr_c, C);
  k_scan_excl<<<1, 256, 0, stream>>>(cnt_v, ptr_v, V);
  k_copy_i32<<<(unsigned)((C + 255) / 256), 256, 0, stream>>>(ptr_c, cnt_c, C);
  k_copy_i32<<<(unsigned)((V + 255) / 256), 256, 0, stream>>>(ptr_v, cnt_v, V);
  k_scatter<<<eBlk, 256, 0, stream>>>(col_idx, cnt_c, perm_c, NNZ);
  k_scatter<<<eBlk, 256, 0, stream>>>(row_idx, cnt_v, perm_v, NNZ);

  // ---- weights: transpose + bf16 ----
  k_transpose_bf16<<<(128 * 64  + 255) / 256, 256, 0, stream>>>(pc_w2,  pc_w2T,  128, 64);
  k_transpose_bf16<<<(128 * 128 + 255) / 256, 256, 0, stream>>>(cu_w1,  cu_w1T,  128, 128);
  k_transpose_bf16<<<(128 * 64  + 255) / 256, 256, 0, stream>>>(cu_w2,  cu_w2T,  128, 64);
  k_transpose_bf16<<<(128 * 128 + 255) / 256, 256, 0, stream>>>(vu_w1,  vu_w1T,  128, 128);
  k_transpose_bf16<<<(128 * 64  + 255) / 256, 256, 0, stream>>>(vu_w2,  vu_w2T,  128, 64);
  k_transpose_bf16<<<(64  * 128 + 255) / 256, 256, 0, stream>>>(out_w1, out_w1T, 64,  128);
  k_transpose_bf16<<<(128 * 16  + 255) / 256, 256, 0, stream>>>(out_w2, out_w2T, 128, 16);

  const long long nV = (long long)V * FM, nC = (long long)C * FM;
  k_fill_f32<<<(unsigned)((nV + 255) / 256), 256, 0, stream>>>(vars_f, nV, 1.0f);
  k_fill_us <<<(unsigned)((nV + 255) / 256), 256, 0, stream>>>(vars_b, nV, (unsigned short)0x3F80);

  const int blkC = (C + 63) / 64, blkV = (V + 63) / 64;
  k_prepare_cond<<<blkC, 128, 0, stream>>>(cond, pc_w1, pc_b1, pc_w2T, pc_b2,
                                           cons_f, cons_b, C);

  const unsigned spmmBlkC = (unsigned)((C + 7) / 8);
  const unsigned spmmBlkV = (unsigned)((V + 7) / 8);
  for (int s = 0; s < PASS_STEPS; ++s) {
    k_spmm_csr<<<spmmBlkC, 256, 0, stream>>>(ptr_c, perm_c, row_idx, adj_vals,
                                             vars_f, msgc_b, C);
    k_mlp2_concat<<<blkC, 128, 0, stream>>>(cons_b, msgc_b, cu_w1T, cu_b1,
                                            cu_w2T, cu_b2, cons_f, cons_b, C);

    k_spmm_csr<<<spmmBlkV, 256, 0, stream>>>(ptr_v, perm_v, col_idx, adj_vals,
                                             cons_f, msgv_b, V);
    k_mlp2_concat<<<blkV, 128, 0, stream>>>(vars_b, msgv_b, vu_w1T, vu_b1,
                                            vu_w2T, vu_b2, vars_f, vars_b, V);
  }

  k_output<<<blkV, 128, 0, stream>>>(vars_b, out_w1T, out_b1, out_w2T, out_b2,
                                     (float*)d_out, V);
}
